// GWNetMix_29429115912857
// MI455X (gfx1250) — matrix operand order repaired and checked by reading, not compiled, not run on hardware
//
#include <hip/hip_runtime.h>
#include <math.h>

// ---------------------------------------------------------------------------
// GWNetMix forward for MI455X (gfx1250). Heavy N x N graph diffusion matmuls
// run on v_wmma_f32_16x16x32_bf16 (32x64 tile / wave, 8 accumulators);
// everything else is fp32 VALU.
// ---------------------------------------------------------------------------

typedef __attribute__((ext_vector_type(16))) __bf16 v16bf;
typedef __attribute__((ext_vector_type(8)))  __bf16 v8bf;
typedef __attribute__((ext_vector_type(8)))  float  v8f;

#define NN      1024          // nodes
#define BB      4             // batch
#define CH      32            // hidden channels
#define NMIX    2
#define NLAY    8
#define SKIPC   256
#define ENDC    512
#define HOR     12
#define NDIM    40
#define TOPK    20

// ============================== small kernels ==============================

__global__ void __launch_bounds__(256) zero_f32_kernel(float* p, long count) {
  long i = (long)blockIdx.x * 256 + threadIdx.x;
  if (i < count) p[i] = 0.f;
}

__global__ void __launch_bounds__(256) f32_to_bf16_kernel(const float* __restrict__ s,
                                                          __bf16* __restrict__ d, long count) {
  long i = (long)blockIdx.x * 256 + threadIdx.x;
  if (i < count) d[i] = (__bf16)s[i];
}

// m = tanh(3 * emb @ lin), both mixes.  tid -> (i, n, e)
__global__ void __launch_bounds__(256) emb_mlp_kernel(const float* __restrict__ emb,
                                                      const float* __restrict__ lin,
                                                      float* __restrict__ out) {
  long tid = (long)blockIdx.x * 256 + threadIdx.x;
  if (tid >= (long)NMIX * NN * NDIM) return;
  int e = tid % NDIM;
  int n = (tid / NDIM) % NN;
  int i = tid / ((long)NDIM * NN);
  const float* er = emb + ((size_t)i * NN + n) * NDIM;
  const float* lc = lin + (size_t)i * NDIM * NDIM + e;
  float acc = 0.f;
  for (int d0 = 0; d0 < NDIM; ++d0) acc += er[d0] * lc[d0 * NDIM];
  out[tid] = tanhf(3.0f * acc);
}

// s[n,m] = sum_d m1[n,d]*m2[m,d]
__global__ void __launch_bounds__(256) sim_gemm_kernel(const float* __restrict__ m1,
                                                       const float* __restrict__ m2,
                                                       float* __restrict__ s) {
  long tid = (long)blockIdx.x * 256 + threadIdx.x;
  if (tid >= (long)NN * NN) return;
  int m = tid & (NN - 1);
  int n = tid >> 10;
  const float* r1 = m1 + (size_t)n * NDIM;
  const float* r2 = m2 + (size_t)m * NDIM;
  float acc = 0.f;
  #pragma unroll 8
  for (int d0 = 0; d0 < NDIM; ++d0) acc += r1[d0] * r2[d0];
  s[tid] = acc;
}

// a[n,m] = relu(tanh(3*(s[n,m]-s[m,n])))
__global__ void __launch_bounds__(256) asym_kernel(const float* __restrict__ s,
                                                   float* __restrict__ a) {
  long tid = (long)blockIdx.x * 256 + threadIdx.x;
  if (tid >= (long)NN * NN) return;
  int m = tid & (NN - 1);
  int n = tid >> 10;
  float v = tanhf(3.0f * (s[(size_t)n * NN + m] - s[(size_t)m * NN + n]));
  a[tid] = fmaxf(v, 0.f);
}

// per-row: keep a[n,m] only if >= 20th largest value in row n
__global__ void __launch_bounds__(256) topk_kernel(float* __restrict__ a) {
  __shared__ float sv[256];
  __shared__ int   si[256];
  const int t = threadIdx.x;
  float* row = a + (size_t)blockIdx.x * NN;
  float v[4];
  #pragma unroll
  for (int q = 0; q < 4; ++q) v[q] = row[t + 256 * q];
  float kth = 0.f;
  for (int k = 0; k < TOPK; ++k) {
    float best = -1e30f; int bq = 0;
    #pragma unroll
    for (int q = 0; q < 4; ++q) if (v[q] > best) { best = v[q]; bq = q; }
    sv[t] = best; si[t] = t * 4 + bq;
    __syncthreads();
    for (int off = 128; off > 0; off >>= 1) {
      if (t < off) {
        if (sv[t + off] > sv[t] || (sv[t + off] == sv[t] && si[t + off] < si[t])) {
          sv[t] = sv[t + off]; si[t] = si[t + off];
        }
      }
      __syncthreads();
    }
    kth = sv[0];
    int widx = si[0];
    if ((widx >> 2) == t) v[widx & 3] = -1e30f;
    __syncthreads();
  }
  #pragma unroll
  for (int q = 0; q < 4; ++q) {
    int m = t + 256 * q;
    float o = row[m];
    row[m] = (o >= kth) ? o : 0.f;
  }
}

// pack fp32 [1024,1024] (optionally transposed) into WMMA-B fragment-major bf16:
// dst[((nt*32 + kt)*32 + lane)*16 + e],  k = kt*32 + (lane/16)*16 + e, col = nt*16 + lane%16
__global__ void __launch_bounds__(256) swizzleB_kernel(const float* __restrict__ src,
                                                       __bf16* __restrict__ dst, int transposed) {
  long tid = (long)blockIdx.x * 256 + threadIdx.x;
  if (tid >= (long)NN * NN) return;
  int e    = tid & 15;
  int lane = (tid >> 4) & 31;
  int kt   = (tid >> 9) & 31;
  int nt   = tid >> 14;
  int col  = lane & 15;
  int k    = kt * 32 + (lane >> 4) * 16 + e;
  int nn   = nt * 16 + col;
  float v  = transposed ? src[(size_t)nn * NN + k] : src[(size_t)k * NN + nn];
  dst[tid] = (__bf16)v;
}

// start 1x1 conv + causal left pad (RF-W = 1):  x[i][b,c,t,n]
__global__ void __launch_bounds__(256) start_kernel(const float* __restrict__ input,
                                                    const float* __restrict__ sw,
                                                    const float* __restrict__ sb,
                                                    float* __restrict__ x0,
                                                    float* __restrict__ x1) {
  long tid = (long)blockIdx.x * 256 + threadIdx.x;
  const long total = (long)NMIX * BB * CH * 13 * NN;
  if (tid >= total) return;
  int n = tid & (NN - 1);
  long r = tid >> 10;
  int t = r % 13; r /= 13;
  int c = r % CH; r /= CH;
  int b = r % BB; r /= BB;
  int i = (int)r;
  float v = 0.f;
  if (t >= 1) v = input[((size_t)(b * 12 + (t - 1)) * NN + n) * NMIX + i];
  float o = sw[i * CH + c] * v + sb[i * CH + c];
  float* dst = (i == 0) ? x0 : x1;
  dst[((size_t)(b * CH + c) * 13 + t) * NN + n] = o;
}

// gated causal dilated conv: x = tanh(filt)*sigmoid(gate)
__global__ void __launch_bounds__(256) dconv_kernel(const float* __restrict__ xin,
                                                    float* __restrict__ xg,
                                                    const float* __restrict__ fw,
                                                    const float* __restrict__ fb,
                                                    const float* __restrict__ gw,
                                                    const float* __restrict__ gb,
                                                    int T_in, int T_out, int d) {
  int n   = blockIdx.x * 256 + threadIdx.x;
  int tau = blockIdx.y;
  int b   = blockIdx.z >> 5;
  int o   = blockIdx.z & 31;
  const float* fwo = fw + o * 64;
  const float* gwo = gw + o * 64;
  float f = fb[o], g = gb[o];
  for (int c = 0; c < CH; ++c) {
    size_t base = ((size_t)(b * CH + c) * T_in) * NN + n;
    float x0 = xin[base + (size_t)tau * NN];
    float x1 = xin[base + (size_t)(tau + d) * NN];
    f += fwo[c * 2] * x0 + fwo[c * 2 + 1] * x1;
    g += gwo[c * 2] * x0 + gwo[c * 2 + 1] * x1;
  }
  float ft = tanhf(f);
  float gs = 1.f / (1.f + expf(-g));
  xg[((size_t)(b * CH + o) * T_out + tau) * NN + n] = ft * gs;
}

// skip conv, last timestep only, accumulate
__global__ void __launch_bounds__(256) skip_kernel(const float* __restrict__ xg,
                                                   float* __restrict__ skip,
                                                   const float* __restrict__ sw,
                                                   const float* __restrict__ sb, int T_out) {
  int n  = blockIdx.x * 256 + threadIdx.x;
  int sc = blockIdx.y;
  int b  = blockIdx.z;
  const float* w = sw + sc * CH;
  float acc = sb[sc];
  size_t tb = (size_t)(T_out - 1) * NN + n;
  for (int c = 0; c < CH; ++c)
    acc += w[c] * xg[((size_t)(b * CH + c) * T_out) * NN + tb];
  skip[((size_t)(b * SKIPC + sc)) * NN + n] += acc;
}

// ============================ WMMA diffusion GEMM ===========================
// P = X * Asw (bf16 in, f32 acc);  Z (+)= alpha * P ;  Pout = bf16(P)
// X: [M,1024] bf16 row-major.  Asw: fragment-major (see swizzleB_kernel).
// 32x64 tile per wave: 2 M-subtiles x 4 N-subtiles = 8 accumulators, each B
// fragment feeds two WMMAs.  alpha = cw[cwIdx] + cwAdd (device scalar).
__global__ void __launch_bounds__(256)
wmma_nconv_kernel(const __bf16* __restrict__ X,
                  const __bf16* __restrict__ Asw,
                  float* __restrict__ Zacc,
                  __bf16* __restrict__ Pout,
                  const float* __restrict__ cw, int cwIdx, float cwAdd,
                  int M, int zeroInit) {
  const int lane = threadIdx.x & 31;
  const int wid  = blockIdx.x * 8 + (threadIdx.x >> 5);
  const int mt   = wid >> 4;        // 32-row M tile
  const int ng   = wid & 15;        // group of 4 N tiles (64 cols)
  if (mt * 32 >= M) return;
  const float alpha = cw[cwIdx] + cwAdd;

  const int row  = lane & 15;
  const int half = lane >> 4;
  const __bf16* xrow0 = X + ((size_t)(mt * 32 + row)) * NN + half * 16;
  const __bf16* xrow1 = xrow0 + (size_t)16 * NN;
  const __bf16* bcol  = Asw + ((size_t)(ng * 4) * 32) * 512 + lane * 16;

  v8f acc[2][4] = {};
  for (int kt = 0; kt < 32; ++kt) {
    v16bf a0, a1;
    {
      const __bf16* xp = xrow0 + kt * 32;
      v8bf lo = *(const v8bf*)(xp);
      v8bf hi = *(const v8bf*)(xp + 8);
      #pragma unroll
      for (int e = 0; e < 8; ++e) { a0[e] = lo[e]; a0[8 + e] = hi[e]; }
    }
    {
      const __bf16* xp = xrow1 + kt * 32;
      v8bf lo = *(const v8bf*)(xp);
      v8bf hi = *(const v8bf*)(xp + 8);
      #pragma unroll
      for (int e = 0; e < 8; ++e) { a1[e] = lo[e]; a1[8 + e] = hi[e]; }
    }
    const __bf16* bp = bcol + (size_t)kt * 512;
    v16bf b0 = *(const v16bf*)(bp);
    v16bf b1 = *(const v16bf*)(bp + 16384);
    v16bf b2 = *(const v16bf*)(bp + 32768);
    v16bf b3 = *(const v16bf*)(bp + 49152);
    acc[0][0] = __builtin_amdgcn_wmma_f32_16x16x32_bf16(false, a0, false, b0, (short)0, acc[0][0], false, false);
    acc[1][0] = __builtin_amdgcn_wmma_f32_16x16x32_bf16(false, a1, false, b0, (short)0, acc[1][0], false, false);
    acc[0][1] = __builtin_amdgcn_wmma_f32_16x16x32_bf16(false, a0, false, b1, (short)0, acc[0][1], false, false);
    acc[1][1] = __builtin_amdgcn_wmma_f32_16x16x32_bf16(false, a1, false, b1, (short)0, acc[1][1], false, false);
    acc[0][2] = __builtin_amdgcn_wmma_f32_16x16x32_bf16(false, a0, false, b2, (short)0, acc[0][2], false, false);
    acc[1][2] = __builtin_amdgcn_wmma_f32_16x16x32_bf16(false, a1, false, b2, (short)0, acc[1][2], false, false);
    acc[0][3] = __builtin_amdgcn_wmma_f32_16x16x32_bf16(false, a0, false, b3, (short)0, acc[0][3], false, false);
    acc[1][3] = __builtin_amdgcn_wmma_f32_16x16x32_bf16(false, a1, false, b3, (short)0, acc[1][3], false, false);
  }
  // C/D layout: VGPR r <-> M = r + 8*(lane/16), N = lane%16
  const int mbase = mt * 32 + (lane >> 4) * 8;
  const int nbase = ng * 64 + (lane & 15);
  #pragma unroll
  for (int h = 0; h < 2; ++h) {
    #pragma unroll
    for (int s = 0; s < 4; ++s) {
      #pragma unroll
      for (int r = 0; r < 8; ++r) {
        size_t idx = (size_t)(mbase + h * 16 + r) * NN + nbase + s * 16;
        float v = acc[h][s][r];
        float z = alpha * v;
        if (!zeroInit) z += Zacc[idx];
        Zacc[idx] = z;
        Pout[idx] = (__bf16)v;
      }
    }
  }
}

// combine: newx = Wg*[Z0|Z1|Z2] + bias*biasTot + residual
__global__ void __launch_bounds__(256)
combine_kernel(const float* __restrict__ Z1, const float* __restrict__ Z2,
               const float* __restrict__ xg_i, const float* __restrict__ xg_j,
               const float* __restrict__ resid,
               const float* __restrict__ Wg, const float* __restrict__ gb,
               const float* __restrict__ cw, int i,
               float* __restrict__ xnext, int T_in, int T_out) {
  int n   = blockIdx.x * 256 + threadIdx.x;
  int tau = blockIdx.y;
  int b   = blockIdx.z >> 5;
  int o   = blockIdx.z & 31;
  int j = 1 - i;
  float c_self  = cw[i * 2 + i] + 1.0f;
  float w_cross = cw[i * 2 + j];
  float coef_i  = 3.0f * c_self;      // 2*(cw_ii+1) + (cw_ii+1)
  float coef_j  = w_cross;
  float biasTot = 3.0f * c_self + w_cross;
  const float* wr = Wg + o * 96;
  float acc = gb[o] * biasTot;
  for (int c = 0; c < CH; ++c) {
    size_t mi = ((size_t)(b * CH + c) * T_out + tau) * NN + n;
    float z0 = coef_i * xg_i[mi] + coef_j * xg_j[mi];
    acc += wr[c] * z0 + wr[32 + c] * Z1[mi] + wr[64 + c] * Z2[mi];
  }
  acc += resid[((size_t)(b * CH + o) * T_in + (T_in - T_out) + tau) * NN + n];
  xnext[((size_t)(b * CH + o) * T_out + tau) * NN + n] = acc;
}

__global__ void __launch_bounds__(256) bn_stats_kernel(const float* __restrict__ x,
                                                       float* __restrict__ stats, int T) {
  __shared__ float sh[256], sh2[256];
  const int o = blockIdx.x;
  const int count = BB * T * NN;
  float s = 0.f, s2 = 0.f;
  for (int idx = threadIdx.x; idx < count; idx += 256) {
    int b = idx / (T * NN);
    int r = idx - b * T * NN;
    float v = x[((size_t)(b * CH + o) * T) * NN + r];
    s += v; s2 += v * v;
  }
  sh[threadIdx.x] = s; sh2[threadIdx.x] = s2;
  __syncthreads();
  for (int off = 128; off > 0; off >>= 1) {
    if (threadIdx.x < off) { sh[threadIdx.x] += sh[threadIdx.x + off]; sh2[threadIdx.x] += sh2[threadIdx.x + off]; }
    __syncthreads();
  }
  if (threadIdx.x == 0) {
    float m = sh[0] / count;
    stats[o * 2] = m;
    stats[o * 2 + 1] = sh2[0] / count - m * m;
  }
}

__global__ void __launch_bounds__(256) bn_apply_kernel(float* __restrict__ x,
                                                       const float* __restrict__ g,
                                                       const float* __restrict__ bb,
                                                       const float* __restrict__ stats,
                                                       int T, long count) {
  long i = (long)blockIdx.x * 256 + threadIdx.x;
  if (i >= count) return;
  int o = (int)((i >> 10) / T) & 31;
  float m = stats[o * 2], v = stats[o * 2 + 1];
  x[i] = g[o] * (x[i] - m) * rsqrtf(v + 1e-5f) + bb[o];
}

__global__ void __launch_bounds__(256) end1_kernel(const float* __restrict__ skipb,
                                                   const float* __restrict__ w,
                                                   const float* __restrict__ bb,
                                                   float* __restrict__ h1) {
  int n = blockIdx.x * 256 + threadIdx.x;
  int e = blockIdx.y;
  int b = blockIdx.z;
  const float* wr = w + e * SKIPC;
  float acc = bb[e];
  for (int sc = 0; sc < SKIPC; ++sc)
    acc += wr[sc] * fmaxf(skipb[((size_t)(b * SKIPC + sc)) * NN + n], 0.f);
  h1[((size_t)(b * ENDC + e)) * NN + n] = fmaxf(acc, 0.f);
}

__global__ void __launch_bounds__(256) end2_kernel(const float* __restrict__ h1,
                                                   const float* __restrict__ w,
                                                   const float* __restrict__ bb,
                                                   float* __restrict__ out, int i) {
  int n = blockIdx.x * 256 + threadIdx.x;
  int h = blockIdx.y;
  int b = blockIdx.z;
  const float* wr = w + h * ENDC;
  float acc = bb[h];
  for (int e = 0; e < ENDC; ++e)
    acc += wr[e] * h1[((size_t)(b * ENDC + e)) * NN + n];
  out[(((size_t)(b * HOR + h)) * NN + n) * NMIX + i] = acc;
}

// ================================= driver ==================================

extern "C" void kernel_launch(void* const* d_in, const int* in_sizes, int n_in,
                              void* d_out, int out_size, void* d_ws, size_t ws_size,
                              hipStream_t stream) {
  const float* in_input = (const float*)d_in[0];
  const float* in_adj   = (const float*)d_in[1];
  const float* in_cw    = (const float*)d_in[2];
  const float* in_emb1  = (const float*)d_in[3];
  const float* in_emb2  = (const float*)d_in[4];
  const float* in_lin1  = (const float*)d_in[5];
  const float* in_lin2  = (const float*)d_in[6];
  const float* in_sw    = (const float*)d_in[7];
  const float* in_sb    = (const float*)d_in[8];
  const float* in_fw    = (const float*)d_in[9];
  const float* in_fb    = (const float*)d_in[10];
  const float* in_gw    = (const float*)d_in[11];
  const float* in_gb    = (const float*)d_in[12];
  const float* in_skw   = (const float*)d_in[13];
  const float* in_skb   = (const float*)d_in[14];
  const float* in_bng   = (const float*)d_in[15];
  const float* in_bnb   = (const float*)d_in[16];
  const float* in_gcw   = (const float*)d_in[17];
  const float* in_gcb   = (const float*)d_in[18];
  const float* in_e1w   = (const float*)d_in[19];
  const float* in_e1b   = (const float*)d_in[20];
  const float* in_e2w   = (const float*)d_in[21];
  const float* in_e2b   = (const float*)d_in[22];
  float* out = (float*)d_out;

  // ---- workspace layout ----
  uint8_t* base = (uint8_t*)d_ws;
  size_t off = 0;
  auto alloc = [&](size_t bytes) -> uint8_t* {
    uint8_t* p = base + off;
    off = (off + bytes + 255) & ~(size_t)255;
    return p;
  };
  const size_t MAT_E = (size_t)NN * NN;                 // 1M elems
  __bf16* ADJSW = (__bf16*)alloc(8 * MAT_E * sizeof(__bf16));   // 8 swizzled graphs
  float*  M1    = (float*)alloc((size_t)NMIX * NN * NDIM * 4);
  float*  M2    = (float*)alloc((size_t)NMIX * NN * NDIM * 4);
  float*  STMP  = (float*)alloc(MAT_E * 4);
  float*  ATMP  = (float*)alloc(MAT_E * 4);
  const size_t XSZ = (size_t)BB * CH * 13 * NN;         // per-mix activation (f32)
  float*  XP[2][2];
  for (int p = 0; p < 2; ++p)
    for (int i = 0; i < NMIX; ++i) XP[p][i] = (float*)alloc(XSZ * 4);
  const size_t XGSZ = (size_t)BB * CH * 12 * NN;
  float*  XG[2];
  for (int i = 0; i < NMIX; ++i) XG[i] = (float*)alloc(XGSZ * 4);
  const size_t MMAX = (size_t)BB * CH * 12;             // max GEMM M = 1536
  __bf16* XGBF[2];
  for (int i = 0; i < NMIX; ++i) XGBF[i] = (__bf16*)alloc(MMAX * NN * sizeof(__bf16));
  __bf16* PBF  = (__bf16*)alloc(MMAX * NN * sizeof(__bf16));
  __bf16* P2BF = (__bf16*)alloc(MMAX * NN * sizeof(__bf16));
  float*  Z1   = (float*)alloc(MMAX * NN * 4);
  float*  Z2   = (float*)alloc(MMAX * NN * 4);
  float*  SKIP = (float*)alloc((size_t)NMIX * BB * SKIPC * NN * 4);
  float*  H1   = (float*)alloc((size_t)BB * ENDC * NN * 4);
  float*  BNST = (float*)alloc(CH * 2 * 4);

  const dim3 blk(256);
  auto gridFor = [](long count) { return dim3((unsigned)((count + 255) / 256)); };

  // ---- setup: zero skip accumulators ----
  {
    long c = (long)NMIX * BB * SKIPC * NN;
    zero_f32_kernel<<<gridFor(c), blk, 0, stream>>>(SKIP, c);
  }

  // ---- adaptive graphs ----
  emb_mlp_kernel<<<gridFor((long)NMIX * NN * NDIM), blk, 0, stream>>>(in_emb1, in_lin1, M1);
  emb_mlp_kernel<<<gridFor((long)NMIX * NN * NDIM), blk, 0, stream>>>(in_emb2, in_lin2, M2);

  // adjacency swizzles: slots {A0, A0^T, A1, A1^T}
  for (int i = 0; i < NMIX; ++i) {
    swizzleB_kernel<<<gridFor(MAT_E), blk, 0, stream>>>(in_adj + i * MAT_E, ADJSW + (size_t)(i * 2) * MAT_E, 0);
    swizzleB_kernel<<<gridFor(MAT_E), blk, 0, stream>>>(in_adj + i * MAT_E, ADJSW + (size_t)(i * 2 + 1) * MAT_E, 1);
  }
  // graph slots 4..7 = G[i][j]
  for (int i = 0; i < NMIX; ++i)
    for (int j = 0; j < NMIX; ++j) {
      sim_gemm_kernel<<<gridFor(MAT_E), blk, 0, stream>>>(M1 + (size_t)i * NN * NDIM,
                                                          M2 + (size_t)j * NN * NDIM, STMP);
      asym_kernel<<<gridFor(MAT_E), blk, 0, stream>>>(STMP, ATMP);
      topk_kernel<<<dim3(NN), blk, 0, stream>>>(ATMP);
      swizzleB_kernel<<<gridFor(MAT_E), blk, 0, stream>>>(ATMP, ADJSW + (size_t)(4 + i * 2 + j) * MAT_E, 0);
    }

  // ---- start conv ----
  start_kernel<<<gridFor((long)NMIX * BB * CH * 13 * NN), blk, 0, stream>>>(
      in_input, in_sw, in_sb, XP[0][0], XP[0][1]);

  // ---- layer loop ----
  const int Tin[NLAY]  = {13, 12, 10, 9, 7, 6, 4, 3};
  const int dil[NLAY]  = {1, 2, 1, 2, 1, 2, 1, 2};
  int cur = 0;
  for (int l = 0; l < NLAY; ++l) {
    const int Ti = Tin[l], d = dil[l], To = Ti - d;
    const int M = BB * CH * To;                     // GEMM rows (multiple of 32)
    const long MN = (long)M * NN;

    for (int i = 0; i < NMIX; ++i) {
      dconv_kernel<<<dim3(4, To, BB * CH), blk, 0, stream>>>(
          XP[cur][i], XG[i],
          in_fw + (size_t)(i * NLAY + l) * 2048, in_fb + (size_t)(i * NLAY + l) * 32,
          in_gw + (size_t)(i * NLAY + l) * 2048, in_gb + (size_t)(i * NLAY + l) * 32,
          Ti, To, d);
      skip_kernel<<<dim3(4, SKIPC, BB), blk, 0, stream>>>(
          XG[i], SKIP + (size_t)i * BB * SKIPC * NN,
          in_skw + (size_t)(i * NLAY + l) * SKIPC * CH,
          in_skb + (size_t)(i * NLAY + l) * SKIPC, To);
      f32_to_bf16_kernel<<<gridFor(MN), blk, 0, stream>>>(XG[i], XGBF[i], MN);
    }

    for (int i = 0; i < NMIX; ++i) {
      // 4 chains: A_i, A_i^T (scale cw_ii+1), G_i0 (src x0), G_i1 (src x1)
      const int   slots[4] = {i * 2, i * 2 + 1, 4 + i * 2 + 0, 4 + i * 2 + 1};
      const int   srcs[4]  = {i, i, 0, 1};
      const int   cwIdx[4] = {i * 2 + i, i * 2 + i, i * 2 + 0, i * 2 + 1};
      const float cwAdd[4] = {1.f, 1.f, (i == 0) ? 1.f : 0.f, (i == 1) ? 1.f : 0.f};
      // waves = (M/32)*16 = 64*To; 8 waves/block -> 8*To blocks
      const dim3 ggrid((unsigned)(8 * To));
      for (int c = 0; c < 4; ++c) {
        const __bf16* A = ADJSW + (size_t)slots[c] * MAT_E;
        wmma_nconv_kernel<<<ggrid, blk, 0, stream>>>(XGBF[srcs[c]], A, Z1, PBF,
                                                     in_cw, cwIdx[c], cwAdd[c], M, c == 0);
        wmma_nconv_kernel<<<ggrid, blk, 0, stream>>>(PBF, A, Z2, P2BF,
                                                     in_cw, cwIdx[c], cwAdd[c], M, c == 0);
      }
      combine_kernel<<<dim3(4, To, BB * CH), blk, 0, stream>>>(
          Z1, Z2, XG[i], XG[1 - i], XP[cur][i],
          in_gcw + (size_t)i * CH * 96, in_gcb + (size_t)i * CH,
          in_cw, i, XP[1 - cur][i], Ti, To);
      bn_stats_kernel<<<dim3(CH), blk, 0, stream>>>(XP[1 - cur][i], BNST, To);
      bn_apply_kernel<<<gridFor(MN), blk, 0, stream>>>(
          XP[1 - cur][i], in_bng + (size_t)(i * NLAY + l) * 32,
          in_bnb + (size_t)(i * NLAY + l) * 32, BNST, To, MN);
    }
    cur ^= 1;
  }

  // ---- output MLPs ----
  for (int i = 0; i < NMIX; ++i) {
    end1_kernel<<<dim3(4, ENDC, BB), blk, 0, stream>>>(
        SKIP + (size_t)i * BB * SKIPC * NN,
        in_e1w + (size_t)i * ENDC * SKIPC, in_e1b + (size_t)i * ENDC, H1);
    end2_kernel<<<dim3(4, HOR, BB), blk, 0, stream>>>(
        H1, in_e2w + (size_t)i * HOR * ENDC, in_e2b + (size_t)i * HOR, out, i);
  }
}
